// GAT_7541962572404
// MI455X (gfx1250) — matrix operand order repaired and checked by reading, not compiled, not run on hardware
//
#include <hip/hip_runtime.h>
#include <hip/hip_bf16.h>
#include <math.h>

typedef __attribute__((ext_vector_type(16))) _Float16 v16h;
typedef __attribute__((ext_vector_type(8)))  _Float16 v8h;
typedef __attribute__((ext_vector_type(8)))  float    v8f;

// ---------- helpers ----------
__device__ __forceinline__ unsigned mapf(float f) {
  unsigned u = __float_as_uint(f);
  return (u & 0x80000000u) ? ~u : (u | 0x80000000u);
}
__device__ __forceinline__ float unmapf(unsigned u) {
  unsigned b = (u & 0x80000000u) ? (u & 0x7FFFFFFFu) : ~u;
  return __uint_as_float(b);
}
__device__ __forceinline__ void edge_sd(const int* ei, int e, int E, int& s, int& d) {
  if (e < E) { s = ei[e]; d = ei[E + e]; }
  else       { s = e - E; d = e - E; }           // self-loop
}

// ---------- conversion / padding kernels (run once per layer, streaming) ----------
// Ah[n, Kp] row-major f16, zero-padded K
__global__ void k_cvt_A(const float* __restrict__ A, _Float16* __restrict__ Ah,
                        int n, int m, int Kp) {
  int idx = blockIdx.x * blockDim.x + threadIdx.x;
  if (idx >= n * Kp) return;
  int row = idx / Kp, c = idx - row * Kp;
  Ah[idx] = (c < m) ? (_Float16)A[(size_t)row * m + c] : (_Float16)0.0f;
}
// Wh[k, Kp] = W[kk, col] transposed to col-major f16, zero-padded K
__global__ void k_cvt_W(const float* __restrict__ W, _Float16* __restrict__ Wh,
                        int m, int k, int Kp) {
  int idx = blockIdx.x * blockDim.x + threadIdx.x;
  if (idx >= k * Kp) return;
  int col = idx / Kp, kk = idx - col * Kp;
  Wh[idx] = (kk < m) ? (_Float16)W[(size_t)kk * k + col] : (_Float16)0.0f;
}

// ---------- WMMA GEMM: C[n,k] = Ah[n,Kp] @ Wh^T  (f16 MACs, f32 acc) ----------
// One wave per 16x64 output strip (4 WMMA tiles): A fragment loaded once per
// k-step and reused across 4 v_wmma issues. Requires k % 64 == 0.
__global__ void k_gemm_wmma(const _Float16* __restrict__ Ah,   // [n, Kp] row-major
                            const _Float16* __restrict__ Wh,   // [k, Kp] col-major
                            float* __restrict__ C, int n, int Kp, int k) {
  int waveId  = (blockIdx.x * blockDim.x + threadIdx.x) >> 5;
  int stripsN = k >> 6;                           // 4 col-tiles per strip
  int strips  = (n >> 4) * stripsN;
  if (waveId >= strips) return;                   // wave-uniform guard
  int sM = waveId / stripsN;
  int sN = waveId - sM * stripsN;
  int lane = threadIdx.x & 31;
  int hsel = lane >> 4;                           // 0: lanes 0-15, 1: lanes 16-31
  int l15  = lane & 15;
  // A frag: row = sM*16+l15, K = k0 + hsel*16 .. +15 (contiguous)
  const _Float16* Ap = Ah + (size_t)((sM << 4) + l15) * Kp + hsel * 16;
  // B frags: col = sN*64 + t*16 + l15, K = k0 + hsel*16 .. +15 (contiguous)
  const _Float16* Bp = Wh + (size_t)((sN << 6) + l15) * Kp + hsel * 16;
  size_t cs = (size_t)16 * Kp;                    // 16-column stride in Wh
  v8f acc0 = {}, acc1 = {}, acc2 = {}, acc3 = {};
  for (int k0 = 0; k0 < Kp; k0 += 32) {
    v8h a0 = *(const v8h*)(Ap + k0);              // 16B
    v8h a1 = *(const v8h*)(Ap + k0 + 8);          // 16B
    v16h a;
#pragma unroll
    for (int e = 0; e < 8; ++e) { a[e] = a0[e]; a[8 + e] = a1[e]; }
    v16h b0 = *(const v16h*)(Bp + k0);            // 32B each (2x b128)
    v16h b1 = *(const v16h*)(Bp + cs + k0);
    v16h b2 = *(const v16h*)(Bp + 2 * cs + k0);
    v16h b3 = *(const v16h*)(Bp + 3 * cs + k0);
    acc0 = __builtin_amdgcn_wmma_f32_16x16x32_f16(false, a, false, b0, (short)0, acc0, false, false);
    acc1 = __builtin_amdgcn_wmma_f32_16x16x32_f16(false, a, false, b1, (short)0, acc1, false, false);
    acc2 = __builtin_amdgcn_wmma_f32_16x16x32_f16(false, a, false, b2, (short)0, acc2, false, false);
    acc3 = __builtin_amdgcn_wmma_f32_16x16x32_f16(false, a, false, b3, (short)0, acc3, false, false);
  }
  // C 16x16 f32 layout per tile: lane group hsel holds rows hsel*8 + v, col = l15
  int row0 = (sM << 4) + hsel * 8;
  int col0 = (sN << 6) + l15;
#pragma unroll
  for (int v = 0; v < 8; ++v) {
    size_t r = (size_t)(row0 + v) * k + col0;
    C[r +  0] = acc0[v];
    C[r + 16] = acc1[v];
    C[r + 32] = acc2[v];
    C[r + 48] = acc3[v];
  }
}

// ---------- attention logits: es/ed[n,H] = sum_c h[n,h,c]*a{s,d}[h,c] ----------
__global__ void k_attn(const float* __restrict__ h, const float* __restrict__ as_,
                       const float* __restrict__ ad_, float* __restrict__ es,
                       float* __restrict__ ed, int n, int H, int C) {
  int gw = (blockIdx.x * blockDim.x + threadIdx.x) >> 5;
  if (gw >= n * H) return;
  int lane = threadIdx.x & 31;
  int node = gw / H, hd = gw - node * H;
  const float* hp = h + (size_t)node * H * C + (size_t)hd * C;
  float ss = 0.f, sd = 0.f;
  for (int c = lane; c < C; c += 32) {
    float v = hp[c];
    ss += v * as_[hd * C + c];
    sd += v * ad_[hd * C + c];
  }
#pragma unroll
  for (int off = 16; off > 0; off >>= 1) {
    ss += __shfl_down(ss, off, 32);
    sd += __shfl_down(sd, off, 32);
  }
  if (lane == 0) { es[gw] = ss; ed[gw] = sd; }
}

// ---------- segment max over incoming edges (mapped-uint atomic max) ----------
__global__ void k_edge_max(const int* __restrict__ ei, const float* __restrict__ es,
                           const float* __restrict__ ed, unsigned* __restrict__ maxb,
                           int E, int Etot, int H) {
  int idx = blockIdx.x * blockDim.x + threadIdx.x;
  if (idx >= Etot * H) return;
  int e = idx / H, hd = idx - e * H;
  int s, d; edge_sd(ei, e, E, s, d);
  float l = es[s * H + hd] + ed[d * H + hd];
  l = (l > 0.f) ? l : 0.2f * l;                   // leaky_relu 0.2
  atomicMax(&maxb[d * H + hd], mapf(l));
}

// ---------- p = exp(e - max[dst]); denom[dst] += p ----------
__global__ void k_edge_exp(const int* __restrict__ ei, const float* __restrict__ es,
                           const float* __restrict__ ed, const unsigned* __restrict__ maxb,
                           float* __restrict__ pbuf, float* __restrict__ denom,
                           int E, int Etot, int H) {
  int idx = blockIdx.x * blockDim.x + threadIdx.x;
  if (idx >= Etot * H) return;
  int e = idx / H, hd = idx - e * H;
  int s, d; edge_sd(ei, e, E, s, d);
  float l = es[s * H + hd] + ed[d * H + hd];
  l = (l > 0.f) ? l : 0.2f * l;
  float m = unmapf(maxb[d * H + hd]);             // always finite (self-loops)
  float pe = expf(l - m);
  pbuf[idx] = pe;
  atomicAdd(&denom[d * H + hd], pe);
}

// ---------- out[dst] += alpha * h[src]  (one wave per edge, float4 gather) ----------
__global__ void k_agg(const int* __restrict__ ei, const float* __restrict__ h,
                      const float* __restrict__ pbuf, const float* __restrict__ denom,
                      float* __restrict__ out, int E, int Etot, int H, int C) {
  int gw = (blockIdx.x * blockDim.x + threadIdx.x) >> 5;
  if (gw >= Etot) return;
  int lane = threadIdx.x & 31;
  int s, d; edge_sd(ei, gw, E, s, d);
  int D_ = H * C;
  const float4* hs = (const float4*)(h + (size_t)s * D_);
  float*        od = out + (size_t)d * D_;
  int nv = D_ >> 2;                                // 64 (layers 1-2) or 16 (layer 3)
  for (int q = lane; q < nv; q += 32) {
    int hd = (q << 2) / C;
    float coef = pbuf[gw * H + hd] / (denom[d * H + hd] + 1e-16f);
    float4 hv = hs[q];
    atomicAdd(od + 4 * q + 0, coef * hv.x);
    atomicAdd(od + 4 * q + 1, coef * hv.y);
    atomicAdd(od + 4 * q + 2, coef * hv.z);
    atomicAdd(od + 4 * q + 3, coef * hv.w);
  }
}

// ---------- x = elu(bn(x + bias)) in place ----------
__global__ void k_bn_elu(float* __restrict__ x, const float* __restrict__ b,
                         const float* __restrict__ g, const float* __restrict__ be,
                         const float* __restrict__ m, const float* __restrict__ v,
                         int n, int d) {
  int idx = blockIdx.x * blockDim.x + threadIdx.x;
  if (idx >= n * d) return;
  int c = idx % d;
  float y = (x[idx] + b[c] - m[c]) * (g[c] * rsqrtf(v[c] + 1e-5f)) + be[c];
  x[idx] = (y > 0.f) ? y : (expf(y) - 1.0f);
}

// ---------- global mean/max pooling accumulators ----------
__global__ void k_pool(const float* __restrict__ h3, const int* __restrict__ batch,
                       float* __restrict__ sum, unsigned* __restrict__ mx,
                       int* __restrict__ cnt, int n, int d) {
  int idx = blockIdx.x * blockDim.x + threadIdx.x;
  if (idx >= n * d) return;
  int node = idx / d, c = idx - node * d;
  int b = batch[node];
  float v = h3[idx];
  atomicAdd(&sum[b * d + c], v);
  atomicMax(&mx[b * d + c], mapf(v));
  if (c == 0) atomicAdd(&cnt[b], 1);
}

// ---------- classifier: relu(pooled @ Wc1 + bc1) @ Wc2 + bc2, one block per graph ----------
__global__ void k_classifier(const float* __restrict__ sum, const unsigned* __restrict__ mx,
                             const int* __restrict__ cnt, const float* __restrict__ Wc1,
                             const float* __restrict__ bc1, const float* __restrict__ Wc2,
                             const float* __restrict__ bc2, float* __restrict__ out,
                             int HID) {
  __shared__ float pooled[128];
  __shared__ float hc[64];
  int b = blockIdx.x, t = threadIdx.x;
  int c_ = cnt[b];
  if (t < HID) {
    pooled[t] = sum[b * HID + t] / (float)max(c_, 1);
  } else if (t < 2 * HID) {
    float f = unmapf(mx[b * HID + (t - HID)]);
    pooled[t] = (f > -INFINITY && f < INFINITY) ? f : 0.0f;   // where(isfinite, f, 0)
  }
  __syncthreads();
  if (t < HID) {
    float acc = bc1[t];
    for (int i = 0; i < 2 * HID; ++i) acc += pooled[i] * Wc1[i * HID + t];
    hc[t] = fmaxf(acc, 0.0f);
  }
  __syncthreads();
  if (t < 2) {
    float acc = bc2[t];
    for (int j = 0; j < HID; ++j) acc += hc[j] * Wc2[j * 2 + t];
    out[b * 2 + t] = acc;
  }
}

// ---------- host-side layer driver ----------
static void run_gat_layer(const float* in, int inDim,
                          const float* W, const float* as_, const float* ad_,
                          const float* b_, const float* g_, const float* be_,
                          const float* mm_, const float* vv_,
                          int H, int C, const int* ei, int E, int N,
                          float* h, float* agg, float* es, float* ed,
                          unsigned* maxb, float* denom, float* pbuf,
                          _Float16* Ah, _Float16* Wh,
                          hipStream_t stream) {
  int outD = H * C;
  int Etot = E + N;
  int Kp   = (inDim + 31) & ~31;
  // 0) convert + pad operands to f16
  {
    long ta = (long)N * Kp;
    k_cvt_A<<<(unsigned)((ta + 255) / 256), 256, 0, stream>>>(in, Ah, N, inDim, Kp);
    long tw = (long)outD * Kp;
    k_cvt_W<<<(unsigned)((tw + 255) / 256), 256, 0, stream>>>(W, Wh, inDim, outD, Kp);
  }
  // 1) node-feature GEMM (WMMA, 16x64 strip per wave)
  int strips = (N / 16) * (outD / 64);
  k_gemm_wmma<<<(strips + 7) / 8, 256, 0, stream>>>(Ah, Wh, h, N, Kp, outD);
  // 2) attention logits
  {
    long waves = (long)N * H;
    k_attn<<<(unsigned)((waves * 32 + 255) / 256), 256, 0, stream>>>(h, as_, ad_, es, ed, N, H, C);
  }
  // 3) zero accumulators
  hipMemsetAsync(maxb, 0, (size_t)N * H * 4, stream);           // mapped 0 == "-inf"
  hipMemsetAsync(denom, 0, (size_t)N * H * 4, stream);
  hipMemsetAsync(agg, 0, (size_t)N * outD * 4, stream);
  // 4) softmax passes + aggregation
  {
    long t1 = (long)Etot * H;
    k_edge_max<<<(unsigned)((t1 + 255) / 256), 256, 0, stream>>>(ei, es, ed, maxb, E, Etot, H);
    k_edge_exp<<<(unsigned)((t1 + 255) / 256), 256, 0, stream>>>(ei, es, ed, maxb, pbuf, denom, E, Etot, H);
    long t2 = (long)Etot * 32;
    k_agg<<<(unsigned)((t2 + 255) / 256), 256, 0, stream>>>(ei, h, pbuf, denom, agg, E, Etot, H, C);
  }
  // 5) bias + BN + ELU
  {
    long t = (long)N * outD;
    k_bn_elu<<<(unsigned)((t + 255) / 256), 256, 0, stream>>>(agg, b_, g_, be_, mm_, vv_, N, outD);
  }
}

extern "C" void kernel_launch(void* const* d_in, const int* in_sizes, int n_in,
                              void* d_out, int out_size, void* d_ws, size_t ws_size,
                              hipStream_t stream) {
  const float* x     = (const float*)d_in[0];
  const int*   ei    = (const int*)d_in[1];
  const int*   batch = (const int*)d_in[2];
  const float* W1  = (const float*)d_in[3];
  const float* as1 = (const float*)d_in[4];
  const float* ad1 = (const float*)d_in[5];
  const float* b1  = (const float*)d_in[6];
  const float* g1  = (const float*)d_in[7];
  const float* be1 = (const float*)d_in[8];
  const float* m1  = (const float*)d_in[9];
  const float* v1  = (const float*)d_in[10];
  const float* W2  = (const float*)d_in[11];
  const float* as2 = (const float*)d_in[12];
  const float* ad2 = (const float*)d_in[13];
  const float* b2  = (const float*)d_in[14];
  const float* g2  = (const float*)d_in[15];
  const float* be2 = (const float*)d_in[16];
  const float* m2  = (const float*)d_in[17];
  const float* v2  = (const float*)d_in[18];
  const float* W3  = (const float*)d_in[19];
  const float* as3 = (const float*)d_in[20];
  const float* ad3 = (const float*)d_in[21];
  const float* b3  = (const float*)d_in[22];
  const float* g3  = (const float*)d_in[23];
  const float* be3 = (const float*)d_in[24];
  const float* m3  = (const float*)d_in[25];
  const float* v3  = (const float*)d_in[26];
  const float* Wc1 = (const float*)d_in[27];
  const float* bc1 = (const float*)d_in[28];
  const float* Wc2 = (const float*)d_in[29];
  const float* bc2 = (const float*)d_in[30];
  float* out = (float*)d_out;

  const int N  = in_sizes[2];          // 20000
  const int E  = in_sizes[1] / 2;      // 320000
  const int IN = in_sizes[0] / N;      // 26
  const int H  = 4, C = 64, D = H * C; // 4 heads x 64
  const int HID = 64;
  const int B  = out_size / 2;         // 64 graphs
  const int Etot = E + N;

  // carve workspace, 256B aligned
  char* p_ = (char*)d_ws;
  auto carve = [&](size_t bytes) -> void* {
    void* r = (void*)p_;
    p_ += (bytes + 255) & ~(size_t)255;
    return r;
  };
  float*    h     = (float*)carve((size_t)N * D * 4);
  float*    agg   = (float*)carve((size_t)N * D * 4);
  float*    es    = (float*)carve((size_t)N * H * 4);
  float*    ed    = (float*)carve((size_t)N * H * 4);
  unsigned* maxb  = (unsigned*)carve((size_t)N * H * 4);
  float*    denom = (float*)carve((size_t)N * H * 4);
  float*    pbuf  = (float*)carve((size_t)Etot * H * 4);
  _Float16* Ah    = (_Float16*)carve((size_t)N * D * 2);      // max Kp = 256
  _Float16* Wh    = (_Float16*)carve((size_t)D * D * 2);      // max k*Kp = 256*256
  float*    psum  = (float*)carve((size_t)B * HID * 4);
  unsigned* pmax  = (unsigned*)carve((size_t)B * HID * 4);
  int*      cnt   = (int*)carve((size_t)B * 4);
  (void)n_in; (void)ws_size;

  // layer 1: x[N,26] -> agg[N,256]
  run_gat_layer(x, IN, W1, as1, ad1, b1, g1, be1, m1, v1, H, C, ei, E, N,
                h, agg, es, ed, maxb, denom, pbuf, Ah, Wh, stream);
  // layer 2: agg[N,256] -> agg[N,256] (GEMM consumes agg into h before agg is reset)
  run_gat_layer(agg, D, W2, as2, ad2, b2, g2, be2, m2, v2, H, C, ei, E, N,
                h, agg, es, ed, maxb, denom, pbuf, Ah, Wh, stream);
  // layer 3: agg[N,256] -> agg[N,64] (1 head, concat=False == identity for 1 head)
  run_gat_layer(agg, D, W3, as3, ad3, b3, g3, be3, m3, v3, 1, HID, ei, E, N,
                h, agg, es, ed, maxb, denom, pbuf, Ah, Wh, stream);

  // pooling
  hipMemsetAsync(psum, 0, (size_t)B * HID * 4, stream);
  hipMemsetAsync(pmax, 0, (size_t)B * HID * 4, stream);   // mapped 0 == "-inf"
  hipMemsetAsync(cnt,  0, (size_t)B * 4, stream);
  {
    long t = (long)N * HID;
    k_pool<<<(unsigned)((t + 255) / 256), 256, 0, stream>>>(agg, batch, psum, pmax, cnt, N, HID);
  }
  // classifier -> d_out [B,2]
  k_classifier<<<B, 128, 0, stream>>>(psum, pmax, cnt, Wc1, bc1, Wc2, bc2, out, HID);
}
